// mvulModel_5214090297910
// MI455X (gfx1250) — matrix operand order repaired and checked by reading, not compiled, not run on hardware
//
#include <hip/hip_runtime.h>
#include <hip/hip_bf16.h>
#include <math.h>

typedef __bf16 bf16;
typedef __attribute__((ext_vector_type(16))) __bf16 v16bf;
typedef __attribute__((ext_vector_type(8)))  float  v8f;

#define NN    16384
#define INC   128
#define OUTC  256
#define CONC  384
#define EE    262144
#define BB    16
#define NPG   1024
#define GRU3  768
#define STEPS 6

// ---------------- WMMA fragment loads (CDNA5 16x16x32 bf16 layouts) ----------
// A 16x32: lanes 0-15 row M=l, v0..3 K=0..7, v4..7 K=16..23; lanes 16-31 K+8.
__device__ __forceinline__ v16bf frag_a(const bf16* A, int lda, int row0, int M,
                                        int kb, int lane) {
  int m = row0 + (lane & 15); if (m >= M) m = M - 1;
  int hi = (lane >> 4) << 3;                    // 0 or 8
  const bf16* base = A + (size_t)m * lda + kb;
  union { v16bf v; unsigned int u[8]; } f;
#pragma unroll
  for (int j = 0; j < 4; ++j) f.u[j]     = *(const unsigned int*)(base + hi + 2*j);
#pragma unroll
  for (int j = 0; j < 4; ++j) f.u[4 + j] = *(const unsigned int*)(base + 16 + hi + 2*j);
  return f.v;
}
// B 32x16: lane n = col, lanes 0-15 K=0..7,16..23 (2 per VGPR), lanes 16-31 K+8.
// W stored row-major [N,K]  (B[k][n] == W[n][k]).
__device__ __forceinline__ v16bf frag_b(const bf16* W, int K, int n0, int N,
                                        int kb, int lane) {
  int n = n0 + (lane & 15); if (n >= N) n = N - 1;
  int khi = (lane >> 4) << 3;                   // 0 or 8
  const bf16* base = W + (size_t)n * K + kb + khi;
  union { v16bf v; unsigned int u[8]; } f;
#pragma unroll
  for (int j = 0; j < 8; ++j) f.u[j] = *(const unsigned int*)(base + (j < 4 ? 2*j : 16 + 2*(j - 4)));
  return f.v;
}

// ---------------- Generic batched GEMM:  C = A * W^T (+bias) (+beta*C) ------
// A: [M,K] bf16 (row stride lda, batch stride sA); W: [N,K] bf16 row-major;
// C: [M,N] f32 (ldc, sC).  Block = 128 thr = 4 waves; wave owns 16x64 slab ->
// block tile 64x64.  K-loop processes 64 per trip with explicit two-stage
// ping-pong buffers: prefetch slice k+32 -> buf1, WMMA on buf0, prefetch
// slice k+64 -> buf0, WMMA on buf1.  No register rotation copies; consumed
// fragments are a full stage old so waits are partial, WMMAs issue
// back-to-back.  Requires K % 64 == 0 (true for 256/384 here).
__global__ __launch_bounds__(128) void k_gemm(
    const bf16* __restrict__ A, int lda, long long sA,
    const bf16* __restrict__ W,
    const float* __restrict__ bias, float beta,
    float* __restrict__ C, int ldc, long long sC,
    int M, int N, int K)
{
  int lane = threadIdx.x & 31, wave = threadIdx.x >> 5;
  int row0 = blockIdx.y * 64 + wave * 16;
  int col0 = blockIdx.x * 64;
  if (row0 >= M) return;
  const bf16* Ab = A + (size_t)blockIdx.z * sA;
  float*      Cb = C + (size_t)blockIdx.z * sC;
  int nn = lane & 15, hi8 = (lane >> 4) << 3;

  v8f acc[4];
#pragma unroll
  for (int t = 0; t < 4; ++t)
#pragma unroll
    for (int r = 0; r < 8; ++r) acc[t][r] = 0.f;

  if (beta != 0.f) {
#pragma unroll
    for (int t = 0; t < 4; ++t) {
      int n = col0 + t * 16 + nn;
      if (n < N)
#pragma unroll
        for (int r = 0; r < 8; ++r) {
          int m = row0 + hi8 + r;
          if (m < M) acc[t][r] = Cb[(size_t)m * ldc + n];
        }
    }
  }

  // stage 0 prologue: slice k=0
  v16bf a0 = frag_a(Ab, lda, row0, M, 0, lane);
  v16bf b0_0 = frag_b(W, K, col0,      N, 0, lane);
  v16bf b0_1 = frag_b(W, K, col0 + 16, N, 0, lane);
  v16bf b0_2 = frag_b(W, K, col0 + 32, N, 0, lane);
  v16bf b0_3 = frag_b(W, K, col0 + 48, N, 0, lane);

  for (int kb = 0; kb < K; kb += 64) {
    // prefetch slice kb+32 into stage-1 buffers (always in-range: K%64==0)
    int k1 = kb + 32;
    v16bf a1   = frag_a(Ab, lda, row0, M, k1, lane);
    v16bf b1_0 = frag_b(W, K, col0,      N, k1, lane);
    v16bf b1_1 = frag_b(W, K, col0 + 16, N, k1, lane);
    v16bf b1_2 = frag_b(W, K, col0 + 32, N, k1, lane);
    v16bf b1_3 = frag_b(W, K, col0 + 48, N, k1, lane);

    acc[0] = __builtin_amdgcn_wmma_f32_16x16x32_bf16(false, a0, false, b0_0, (short)0, acc[0], false, false);
    acc[1] = __builtin_amdgcn_wmma_f32_16x16x32_bf16(false, a0, false, b0_1, (short)0, acc[1], false, false);
    acc[2] = __builtin_amdgcn_wmma_f32_16x16x32_bf16(false, a0, false, b0_2, (short)0, acc[2], false, false);
    acc[3] = __builtin_amdgcn_wmma_f32_16x16x32_bf16(false, a0, false, b0_3, (short)0, acc[3], false, false);

    // prefetch slice kb+64 into stage-0 buffers (clamped on last trip)
    int k0 = (kb + 64 < K) ? (kb + 64) : 0;
    a0   = frag_a(Ab, lda, row0, M, k0, lane);
    b0_0 = frag_b(W, K, col0,      N, k0, lane);
    b0_1 = frag_b(W, K, col0 + 16, N, k0, lane);
    b0_2 = frag_b(W, K, col0 + 32, N, k0, lane);
    b0_3 = frag_b(W, K, col0 + 48, N, k0, lane);

    acc[0] = __builtin_amdgcn_wmma_f32_16x16x32_bf16(false, a1, false, b1_0, (short)0, acc[0], false, false);
    acc[1] = __builtin_amdgcn_wmma_f32_16x16x32_bf16(false, a1, false, b1_1, (short)0, acc[1], false, false);
    acc[2] = __builtin_amdgcn_wmma_f32_16x16x32_bf16(false, a1, false, b1_2, (short)0, acc[2], false, false);
    acc[3] = __builtin_amdgcn_wmma_f32_16x16x32_bf16(false, a1, false, b1_3, (short)0, acc[3], false, false);
  }

#pragma unroll
  for (int t = 0; t < 4; ++t) {
    int n = col0 + t * 16 + nn;
    if (n >= N) continue;
    float bv = bias ? bias[n] : 0.f;
#pragma unroll
    for (int r = 0; r < 8; ++r) {
      int m = row0 + hi8 + r;
      if (m < M) Cb[(size_t)m * ldc + n] = acc[t][r] + bv;
    }
  }
}

// ---------------- Elementwise / reduction kernels ---------------------------
__global__ void k_pad(const float* __restrict__ feat, float* __restrict__ h,
                      bf16* __restrict__ hbf, bf16* __restrict__ fbf) {
  long idx = (long)blockIdx.x * blockDim.x + threadIdx.x;   // N*256
  int c = idx & 255; long n = idx >> 8;
  float v = (c < INC) ? feat[n * INC + c] : 0.f;
  h[idx] = v; hbf[idx] = (bf16)v;
  if (c < INC) fbf[n * INC + c] = (bf16)v;
}

__global__ void k_f2bf(const float* __restrict__ a, bf16* __restrict__ o, long n) {
  long i = (long)blockIdx.x * blockDim.x + threadIdx.x;
  if (i < n) o[i] = (bf16)a[i];
}

// [O,I,K] f32 -> [K][O][I] bf16 (K=1 is a plain convert)
__global__ void k_pack_convw(const float* __restrict__ w, bf16* __restrict__ o,
                             int O, int I, int K) {
  long idx = (long)blockIdx.x * blockDim.x + threadIdx.x;
  long total = (long)O * I * K;
  if (idx >= total) return;
  int kk = idx % K; long t = idx / K; int i = t % I; int ot = t / I;
  o[((long)kk * O + ot) * I + i] = (bf16)w[idx];
}

__global__ void k_zero(float* __restrict__ p, long n) {
  long i = (long)blockIdx.x * blockDim.x + threadIdx.x;
  if (i < n) p[i] = 0.f;
}

// one block per edge, 256 channels; agg[dst] += Wh[etype][src]
__global__ void k_scatter(const float* __restrict__ Wh, const int* __restrict__ src,
                          const int* __restrict__ dst, const int* __restrict__ et,
                          float* __restrict__ agg) {
  int e = blockIdx.x, c = threadIdx.x;
  float v = Wh[((long)et[e] * NN + src[e]) * OUTC + c];
  atomicAdd(&agg[(long)dst[e] * OUTC + c], v);
}

__global__ void k_gru(const float* __restrict__ gi, const float* __restrict__ gh,
                      float* __restrict__ h, bf16* __restrict__ hbf) {
  long idx = (long)blockIdx.x * blockDim.x + threadIdx.x;   // N*256
  int c = idx & 255; long n = idx >> 8;
  const float* gin = gi + n * GRU3;
  const float* ghn = gh + n * GRU3;
  float r  = 1.f / (1.f + expf(-(gin[c]        + ghn[c])));
  float z  = 1.f / (1.f + expf(-(gin[256 + c]  + ghn[256 + c])));
  float ng = tanhf(gin[512 + c] + r * ghn[512 + c]);
  float hn = (1.f - z) * ng + z * h[idx];
  h[idx] = hn; hbf[idx] = (bf16)hn;
}

__global__ void k_concat(const bf16* __restrict__ hbf, const bf16* __restrict__ fbf,
                         bf16* __restrict__ cbf) {
  long idx = (long)blockIdx.x * blockDim.x + threadIdx.x;   // N*384
  if (idx >= (long)NN * CONC) return;
  int c = idx % CONC; long n = idx / CONC;
  cbf[idx] = (c < OUTC) ? hbf[n * OUTC + c] : fbf[n * INC + (c - OUTC)];
}

// training-mode BN stats over (B,L) per channel; x: [B,L,C] channels-last
__global__ void k_bn_stats(const float* __restrict__ x, int Bb, int L, int C,
                           float* __restrict__ stats) {
  int c = blockIdx.x;
  long total = (long)Bb * L;
  float s = 0.f, sq = 0.f;
  for (long i = threadIdx.x; i < total; i += blockDim.x) {
    float v = x[i * C + c]; s += v; sq += v * v;
  }
  __shared__ float sh[512];
  sh[threadIdx.x] = s; sh[256 + threadIdx.x] = sq; __syncthreads();
  for (int o = 128; o > 0; o >>= 1) {
    if ((int)threadIdx.x < o) {
      sh[threadIdx.x] += sh[threadIdx.x + o];
      sh[256 + threadIdx.x] += sh[256 + threadIdx.x + o];
    }
    __syncthreads();
  }
  if (threadIdx.x == 0) {
    float m = sh[0] / (float)total;
    stats[c] = m; stats[C + c] = sh[256] / (float)total - m * m;
  }
}

__global__ void k_bn_relu_pool(const float* __restrict__ x, const float* __restrict__ stats,
                               const float* __restrict__ g, const float* __restrict__ bb,
                               float* __restrict__ outf, bf16* __restrict__ outb,
                               int Bb, int L, int C, int k, int s) {
  int Lout = (L - k) / s + 1;
  long idx = (long)blockIdx.x * blockDim.x + threadIdx.x;
  long total = (long)Bb * Lout * C;
  if (idx >= total) return;
  int c = idx % C; long t = idx / C; int lo = t % Lout; int b = t / Lout;
  float sc = rsqrtf(stats[C + c] + 1e-5f) * g[c];
  float bi = bb[c] - stats[c] * sc;
  const float* xr = x + ((long)b * L + (long)lo * s) * C + c;
  float best = -INFINITY;
  for (int j = 0; j < k; ++j) {
    float v = xr[(long)j * C] * sc + bi;
    v = v > 0.f ? v : 0.f;
    best = v > best ? v : best;
  }
  long o = ((long)b * Lout + lo) * C + c;
  if (outf) outf[o] = best;
  if (outb) outb[o] = (bf16)best;
}

// (Y2 @ mlpy^T + by) * (Z2 @ mlpz^T + bz), mean over L=255, sigmoid. 1 block/b.
__global__ void k_final(const float* __restrict__ Y2, const float* __restrict__ Z2,
                        const float* __restrict__ myw, const float* __restrict__ myb,
                        const float* __restrict__ mzw, const float* __restrict__ mzb,
                        float* __restrict__ out) {
  int b = blockIdx.x, t = threadIdx.x;
  float s0 = 0.f, s1 = 0.f;
  for (int l = t; l < 255; l += 256) {
    const float* y = Y2 + ((long)b * 255 + l) * OUTC;
    const float* z = Z2 + ((long)b * 255 + l) * CONC;
    float y0 = myb[0], y1 = myb[1];
    for (int c = 0; c < OUTC; ++c) { float v = y[c]; y0 += v * myw[c]; y1 += v * myw[OUTC + c]; }
    float z0 = mzb[0], z1 = mzb[1];
    for (int c = 0; c < CONC; ++c) { float v = z[c]; z0 += v * mzw[c]; z1 += v * mzw[CONC + c]; }
    s0 += y0 * z0; s1 += y1 * z1;
  }
  __shared__ float sh[512];
  sh[t] = s0; sh[256 + t] = s1; __syncthreads();
  for (int o = 128; o > 0; o >>= 1) {
    if (t < o) { sh[t] += sh[t + o]; sh[256 + t] += sh[256 + t + o]; }
    __syncthreads();
  }
  if (t == 0) {
    out[b * 2 + 0] = 1.f / (1.f + expf(-sh[0]   / 255.f));
    out[b * 2 + 1] = 1.f / (1.f + expf(-sh[256] / 255.f));
  }
}

// ---------------- host orchestration ----------------------------------------
static inline long cdiv(long a, long b) { return (a + b - 1) / b; }

extern "C" void kernel_launch(void* const* d_in, const int* in_sizes, int n_in,
                              void* d_out, int out_size, void* d_ws, size_t ws_size,
                              hipStream_t stream) {
  const float* features = (const float*)d_in[0];
  const int*   src      = (const int*)d_in[1];
  const int*   dst      = (const int*)d_in[2];
  const int*   etype    = (const int*)d_in[3];
  const float* W_lin    = (const float*)d_in[4];
  const float* b_lin    = (const float*)d_in[5];
  const float* W_ih     = (const float*)d_in[6];
  const float* W_hh     = (const float*)d_in[7];
  const float* b_ih     = (const float*)d_in[8];
  const float* b_hh     = (const float*)d_in[9];
  const float* conv1_w  = (const float*)d_in[10];
  const float* conv1_b  = (const float*)d_in[11];
  const float* conv2_w  = (const float*)d_in[12];
  const float* conv2_b  = (const float*)d_in[13];
  const float* convc1_w = (const float*)d_in[14];
  const float* convc1_b = (const float*)d_in[15];
  const float* convc2_w = (const float*)d_in[16];
  const float* convc2_b = (const float*)d_in[17];
  const float* bn_g     = (const float*)d_in[18];
  const float* bn_b     = (const float*)d_in[19];
  const float* bnc_g    = (const float*)d_in[20];
  const float* bnc_b    = (const float*)d_in[21];
  const float* mlpy_w   = (const float*)d_in[22];
  const float* mlpy_b   = (const float*)d_in[23];
  const float* mlpz_w   = (const float*)d_in[24];
  const float* mlpz_b   = (const float*)d_in[25];
  float* out = (float*)d_out;

  // workspace carve-up (phase-aliased)
  char* ws = (char*)d_ws;
  size_t off = 0;
  auto take = [&](size_t bytes) { char* p = ws + off; off += (bytes + 255) & ~(size_t)255; return p; };
  float* h      = (float*)take((size_t)NN * OUTC * 4);          // 16 MB
  bf16*  hbf    = (bf16*) take((size_t)NN * OUTC * 2);          // 8 MB
  bf16*  fbf    = (bf16*) take((size_t)NN * INC * 2);           // 4 MB
  float* Wh     = (float*)take(2ull * NN * OUTC * 4);           // 33.6 MB  (later: Z1raw)
  float* agg    = (float*)take((size_t)NN * OUTC * 4);          // 16.8 MB  (later: Y1raw)
  bf16*  aggbf  = (bf16*) take((size_t)NN * OUTC * 2);          // 8.4 MB   (later: Y1p bf16)
  float* gi     = (float*)take((size_t)NN * GRU3 * 4);          // 50.3 MB  (later: Y2/Z2 stages)
  float* gh     = (float*)take((size_t)NN * GRU3 * 4);          // 50.3 MB  (later: c_bf)
  bf16*  Wlinbf = (bf16*) take(2ull * OUTC * OUTC * 2);
  bf16*  Wihbf  = (bf16*) take((size_t)GRU3 * OUTC * 2);
  bf16*  Whhbf  = (bf16*) take((size_t)GRU3 * OUTC * 2);
  bf16*  c1wbf  = (bf16*) take(3ull * OUTC * OUTC * 2);
  bf16*  c2wbf  = (bf16*) take((size_t)OUTC * OUTC * 2);
  bf16*  cc1wbf = (bf16*) take(3ull * CONC * CONC * 2);
  bf16*  cc2wbf = (bf16*) take((size_t)CONC * CONC * 2);
  float* stats  = (float*)take(2ull * CONC * 4);
  // aliases for the conv tail
  float* Y1raw = agg;                         // [16,1022,256]
  bf16*  Y1pbf = aggbf;                       // [16,510,256]
  float* Z1raw = Wh;                          // [16,1022,384]
  char*  giB   = (char*)gi;
  float* Y2raw = (float*)(giB);               // [16,510,256]
  float* Z2raw = (float*)(giB +  8388608);    // [16,510,384]
  float* Y2    = (float*)(giB + 20971520);    // [16,255,256]
  float* Z2    = (float*)(giB + 25165824);    // [16,255,384]
  bf16*  Z1pbf = (bf16*) (giB + 31457280);    // [16,510,384]
  bf16*  cbf   = (bf16*)gh;                   // [16384,384]

  // --- weight conversion / packing (cheap, deterministic every call) ---
  k_f2bf<<<cdiv(2*OUTC*OUTC,256),256,0,stream>>>(W_lin, Wlinbf, 2*OUTC*OUTC);
  k_f2bf<<<cdiv(GRU3*OUTC,256),256,0,stream>>>(W_ih, Wihbf, (long)GRU3*OUTC);
  k_f2bf<<<cdiv(GRU3*OUTC,256),256,0,stream>>>(W_hh, Whhbf, (long)GRU3*OUTC);
  k_pack_convw<<<cdiv(3*OUTC*OUTC,256),256,0,stream>>>(conv1_w,  c1wbf,  OUTC, OUTC, 3);
  k_pack_convw<<<cdiv(OUTC*OUTC,256),256,0,stream>>>(conv2_w,  c2wbf,  OUTC, OUTC, 1);
  k_pack_convw<<<cdiv(3*CONC*CONC,256),256,0,stream>>>(convc1_w, cc1wbf, CONC, CONC, 3);
  k_pack_convw<<<cdiv(CONC*CONC,256),256,0,stream>>>(convc2_w, cc2wbf, CONC, CONC, 1);

  // --- init h = pad(features) ---
  k_pad<<<NN*OUTC/256,256,0,stream>>>(features, h, hbf, fbf);

  // --- GGNN steps ---
  for (int s = 0; s < STEPS; ++s) {
    dim3 gWh(OUTC/64, NN/64, 1);
    k_gemm<<<gWh,128,0,stream>>>(hbf, OUTC, 0, Wlinbf,              b_lin,        0.f,
                                 Wh,               OUTC, 0, NN, OUTC, OUTC);
    k_gemm<<<gWh,128,0,stream>>>(hbf, OUTC, 0, Wlinbf + OUTC*OUTC,  b_lin + OUTC, 0.f,
                                 Wh + (size_t)NN*OUTC, OUTC, 0, NN, OUTC, OUTC);
    k_zero<<<NN*OUTC/256,256,0,stream>>>(agg, (long)NN*OUTC);
    k_scatter<<<EE,OUTC,0,stream>>>(Wh, src, dst, etype, agg);
    k_f2bf<<<NN*OUTC/256,256,0,stream>>>(agg, aggbf, (long)NN*OUTC);
    dim3 gG(GRU3/64, NN/64, 1);
    k_gemm<<<gG,128,0,stream>>>(aggbf, OUTC, 0, Wihbf, b_ih, 0.f, gi, GRU3, 0, NN, GRU3, OUTC);
    k_gemm<<<gG,128,0,stream>>>(hbf,   OUTC, 0, Whhbf, b_hh, 0.f, gh, GRU3, 0, NN, GRU3, OUTC);
    k_gru<<<NN*OUTC/256,256,0,stream>>>(gi, gh, h, hbf);
  }

  // --- c_i = concat(h, x) (bf16, channels-last) ---
  k_concat<<<cdiv((long)NN*CONC,256),256,0,stream>>>(hbf, fbf, cbf);

  // --- Y branch: conv k3 (3 shifted GEMMs) -> BN/ReLU/pool3s2 -> conv k1 -> BN/ReLU/pool2s2
  for (int k = 0; k < 3; ++k)
    k_gemm<<<dim3(OUTC/64, cdiv(1022,64), BB),128,0,stream>>>(
        hbf + (size_t)k*OUTC, OUTC, (long long)NPG*OUTC,
        c1wbf + (size_t)k*OUTC*OUTC, k ? nullptr : conv1_b, k ? 1.f : 0.f,
        Y1raw, OUTC, (long long)1022*OUTC, 1022, OUTC, OUTC);
  k_bn_stats<<<OUTC,256,0,stream>>>(Y1raw, BB, 1022, OUTC, stats);
  k_bn_relu_pool<<<cdiv((long)BB*510*OUTC,256),256,0,stream>>>(
      Y1raw, stats, bn_g, bn_b, nullptr, Y1pbf, BB, 1022, OUTC, 3, 2);
  k_gemm<<<dim3(OUTC/64, cdiv(510,64), BB),128,0,stream>>>(
      Y1pbf, OUTC, (long long)510*OUTC, c2wbf, conv2_b, 0.f,
      Y2raw, OUTC, (long long)510*OUTC, 510, OUTC, OUTC);
  k_bn_stats<<<OUTC,256,0,stream>>>(Y2raw, BB, 510, OUTC, stats);
  k_bn_relu_pool<<<cdiv((long)BB*255*OUTC,256),256,0,stream>>>(
      Y2raw, stats, bn_g, bn_b, Y2, nullptr, BB, 510, OUTC, 2, 2);

  // --- Z branch (CONCAT channels) ---
  for (int k = 0; k < 3; ++k)
    k_gemm<<<dim3(CONC/64, cdiv(1022,64), BB),128,0,stream>>>(
        cbf + (size_t)k*CONC, CONC, (long long)NPG*CONC,
        cc1wbf + (size_t)k*CONC*CONC, k ? nullptr : convc1_b, k ? 1.f : 0.f,
        Z1raw, CONC, (long long)1022*CONC, 1022, CONC, CONC);
  k_bn_stats<<<CONC,256,0,stream>>>(Z1raw, BB, 1022, CONC, stats);
  k_bn_relu_pool<<<cdiv((long)BB*510*CONC,256),256,0,stream>>>(
      Z1raw, stats, bnc_g, bnc_b, nullptr, Z1pbf, BB, 1022, CONC, 3, 2);
  k_gemm<<<dim3(CONC/64, cdiv(510,64), BB),128,0,stream>>>(
      Z1pbf, CONC, (long long)510*CONC, cc2wbf, convc2_b, 0.f,
      Z2raw, CONC, (long long)510*CONC, 510, CONC, CONC);
  k_bn_stats<<<CONC,256,0,stream>>>(Z2raw, BB, 510, CONC, stats);
  k_bn_relu_pool<<<cdiv((long)BB*255*CONC,256),256,0,stream>>>(
      Z2raw, stats, bnc_g, bnc_b, Z2, nullptr, BB, 510, CONC, 2, 2);

  // --- gating MLPs, mean over L, sigmoid ---
  k_final<<<BB,256,0,stream>>>(Y2, Z2, mlpy_w, mlpy_b, mlpz_w, mlpz_b, out);
}
